// GraphNet_13615046328672
// MI455X (gfx1250) — matrix operand order repaired and checked by reading, not compiled, not run on hardware
//
#include <hip/hip_runtime.h>
#include <stdint.h>

typedef __attribute__((ext_vector_type(16))) _Float16 v16h;
typedef __attribute__((ext_vector_type(8)))  float    v8f;

#define NNODES 25000
#define NEDGES 400000
#define CD 128
#define LN_EPS 1e-5f

// LDS row strides (in halves); chosen so consecutive rows land ~4 banks apart.
constexpr int SA384 = 392;   // 16 x 384 input tiles
constexpr int SH128 = 136;   // 16 x 128 hidden tiles
constexpr int SN256 = 264;   // 16 x 256 node-mlp1 input tile

union ABfrag { unsigned int u[8]; v16h h; };

__device__ __forceinline__ v8f wmma_f16(v16h a, v16h b, v8f c) {
  // D = A(16x32 f16) * B(32x16 f16) + C(16x16 f32)
  return __builtin_amdgcn_wmma_f32_16x16x32_f16(false, a, false, b, (short)0, c,
                                                false, false);
}

// A tile: LDS, 16 rows x K cols f16, row stride SA (halves).
// B: global f16, transposed weights Wt[n][k], row stride K (halves).
// acc[nt] holds the 16x16 f32 D tile for output cols [nt*16, nt*16+16).
template<int K, int NT, int SA>
__device__ __forceinline__ void gemm_tile(const _Float16* __restrict__ A,
                                          const _Float16* __restrict__ Wt,
                                          v8f* acc) {
  const int l  = threadIdx.x & 31;
  const int hh = l >> 4;       // lane half (0: lanes 0-15, 1: lanes 16-31)
  const int nm = l & 15;       // A row / B column within half
  const unsigned int* a32 = (const unsigned int*)A;
  const unsigned int* b32 = (const unsigned int*)Wt;
#pragma unroll
  for (int nt = 0; nt < NT; ++nt)
#pragma unroll
    for (int i = 0; i < 8; ++i) acc[nt][i] = 0.0f;

  for (int kt = 0; kt < K / 32; ++kt) {
    ABfrag a;
#pragma unroll
    for (int v = 0; v < 8; ++v) {
      // ISA 16-bit A 16x32 layout: v<4 -> K = 2v + 8*half ; v>=4 -> 16 + 2(v-4) + 8*half
      const int kb = (v < 4) ? (2 * v + 8 * hh) : (16 + 2 * (v - 4) + 8 * hh);
      a.u[v] = a32[(nm * SA + kt * 32 + kb) >> 1];
    }
#pragma unroll
    for (int nt = 0; nt < NT; ++nt) {
      ABfrag b;
#pragma unroll
      for (int v = 0; v < 8; ++v) {
        // B 32x16 layout matching A: v<4 -> K = 2v + 8*half ; v>=4 -> 16 + 2(v-4) + 8*half, N = lane%16
        const int kb = (v < 4) ? (2 * v + 8 * hh) : (16 + 2 * (v - 4) + 8 * hh);
        b.u[v] = b32[((nt * 16 + nm) * K + kt * 32 + kb) >> 1];
      }
      acc[nt] = wmma_f16(a.h, b.h, acc[nt]);
    }
  }
}

__device__ __forceinline__ float gelu_tanh(float x) {
  return 0.5f * x * (1.0f + tanhf(0.7978845608f * (x + 0.044715f * x * x * x)));
}

// Bias + LayerNorm(axis=-1, 128 wide) + GELU on the accumulator layout,
// emit f16 tile to LDS (row m, col n), row stride SO.
// Row m = r + 8*half lives in acc[*][r] across the 16 lanes of that half,
// so the reduction is per-register partials + 4 xor-shuffles within a half.
template<int NT>
__device__ __forceinline__ void ln_gelu_to_lds(v8f* acc,
    const float* __restrict__ bias, const float* __restrict__ g,
    const float* __restrict__ be, _Float16* __restrict__ out, int SO) {
  const int l = threadIdx.x & 31;
  const int hh = l >> 4, nl = l & 15;
  float s[8], q[8];
#pragma unroll
  for (int r = 0; r < 8; ++r) { s[r] = 0.f; q[r] = 0.f; }
#pragma unroll
  for (int nt = 0; nt < NT; ++nt) {
    const float bv = bias[nt * 16 + nl];
#pragma unroll
    for (int r = 0; r < 8; ++r) {
      float x = acc[nt][r] + bv;
      acc[nt][r] = x;
      s[r] += x; q[r] += x * x;
    }
  }
#pragma unroll
  for (int r = 0; r < 8; ++r) {
#pragma unroll
    for (int msk = 1; msk < 16; msk <<= 1) {   // stays within 16-lane halves
      s[r] += __shfl_xor(s[r], msk, 32);
      q[r] += __shfl_xor(q[r], msk, 32);
    }
  }
  const float invN = 1.0f / (float)(NT * 16);
#pragma unroll
  for (int r = 0; r < 8; ++r) {
    const float mu = s[r] * invN;
    const float var = q[r] * invN - mu * mu;
    s[r] = mu;
    q[r] = rsqrtf(var + LN_EPS);
  }
#pragma unroll
  for (int nt = 0; nt < NT; ++nt) {
    const int n = nt * 16 + nl;
    const float gg = g[n], bb = be[n];
#pragma unroll
    for (int r = 0; r < 8; ++r) {
      const float x = (acc[nt][r] - s[r]) * q[r] * gg + bb;
      out[(r + 8 * hh) * SO + n] = (_Float16)gelu_tanh(x);
    }
  }
}

__device__ __forceinline__ void store4h(_Float16* dst, float4 v) {
  dst[0] = (_Float16)v.x; dst[1] = (_Float16)v.y;
  dst[2] = (_Float16)v.z; dst[3] = (_Float16)v.w;
}

// ---------------------------------------------------------------- prep kernels

__global__ void wt_convert(const float* __restrict__ W, _Float16* __restrict__ Wt,
                           int K, int N) {
  int idx = blockIdx.x * 256 + threadIdx.x;
  if (idx < K * N) {
    int k = idx / N, n = idx - k * N;
    Wt[n * K + k] = (_Float16)W[idx];    // store transposed: Wt[n][k]
  }
}

__global__ void zero_f32(float* __restrict__ p, int n) {
  int i = blockIdx.x * 256 + threadIdx.x;
  if (i < n) p[i] = 0.0f;
}

__global__ void copy_u(const float* __restrict__ u, float* __restrict__ out) {
  if (threadIdx.x < 16) out[threadIdx.x] = u[threadIdx.x];
}

// ---------------------------------------------------------------- edge kernel

struct EdgeParams {
  const int* edge_index; const int* c_ids; const int* rel_ids;
  const float* concept; const float* rel;
  const _Float16 *eW0t, *eW1t, *eW2t, *n1W0t, *n1W1t, *n1W2t;
  const float *eb0, *eg0, *ebe0, *eb1, *eg1, *ebe1, *eb2;
  const float *n1b0, *n1g0, *n1be0, *n1b1, *n1g1, *n1be1, *n1b2;
  float* eaOut; float* sums; float* cnt;
};

__global__ __launch_bounds__(64) void edge_kernel(EdgeParams p) {
  __shared__ __align__(16) _Float16 sA [2][16 * SA384];
  __shared__ __align__(16) _Float16 sH1[2][16 * SH128];
  __shared__ __align__(16) _Float16 sH2[2][16 * SH128];
  __shared__ __align__(16) _Float16 sN [2][16 * SN256];
  __shared__ int sCol[2][16];

  const int w = threadIdx.x >> 5;
  const int l = threadIdx.x & 31;
  const int eBase = (blockIdx.x * 2 + w) * 16;   // 400000/16 = 25000 tiles, no tail

  _Float16* A  = sA[w];
  _Float16* H1 = sH1[w];
  _Float16* H2 = sH2[w];
  _Float16* NB = sN[w];
  int* colArr  = sCol[w];

  // ---- gather: A = [x[row] | x[col] | rel_emb], NB[:,0:128] = x[row] ----
  {
    const int m = l >> 1, pp = l & 1;            // 2 lanes per row, 64 cols each
    const int e  = eBase + m;
    const int r  = p.edge_index[e];
    const int c  = p.edge_index[NEDGES + e];
    if (pp == 0) colArr[m] = c;
    const int rl = p.rel_ids[e];
    const float4* xr = (const float4*)(p.concept + (size_t)p.c_ids[r] * CD);
    const float4* xc = (const float4*)(p.concept + (size_t)p.c_ids[c] * CD);
    const float4* xe = (const float4*)(p.rel + (size_t)rl * CD);
#pragma unroll
    for (int i = 0; i < 16; ++i) {
      const int ci = pp * 64 + i * 4;
      float4 v0 = xr[pp * 16 + i];
      float4 v1 = xc[pp * 16 + i];
      float4 v2 = xe[pp * 16 + i];
      store4h(A + m * SA384 + ci, v0);
      store4h(NB + m * SN256 + ci, v0);
      store4h(A + m * SA384 + 128 + ci, v1);
      store4h(A + m * SA384 + 256 + ci, v2);
    }
  }
  __syncthreads();

  // ---- edge MLP: 384 -> 128 (LN,GELU) -> 128 (LN,GELU) -> 128 ----
  { v8f acc[8]; gemm_tile<384, 8, SA384>(A, p.eW0t, acc);
    ln_gelu_to_lds<8>(acc, p.eb0, p.eg0, p.ebe0, H1, SH128); }
  __syncthreads();
  { v8f acc[8]; gemm_tile<128, 8, SH128>(H1, p.eW1t, acc);
    ln_gelu_to_lds<8>(acc, p.eb1, p.eg1, p.ebe1, H2, SH128); }
  __syncthreads();
  { v8f acc[8]; gemm_tile<128, 8, SH128>(H2, p.eW2t, acc);
    const int hh = l >> 4, nl = l & 15;
#pragma unroll
    for (int nt = 0; nt < 8; ++nt) {
      const int n = nt * 16 + nl;
      const float bv = p.eb2[n];
#pragma unroll
      for (int r = 0; r < 8; ++r) {
        const int m2 = r + 8 * hh;
        const float x = acc[nt][r] + bv;
        p.eaOut[(size_t)(eBase + m2) * CD + n] = x;      // ea output (f32)
        NB[m2 * SN256 + 128 + n] = (_Float16)x;          // node-mlp1 input cols 128..255
      }
    }
  }
  __syncthreads();

  // ---- node MLP1: 256 -> 128 (LN,GELU) -> 128 (LN,GELU) -> 256, scatter ----
  { v8f acc[8]; gemm_tile<256, 8, SN256>(NB, p.n1W0t, acc);
    ln_gelu_to_lds<8>(acc, p.n1b0, p.n1g0, p.n1be0, H1, SH128); }
  __syncthreads();
  { v8f acc[8]; gemm_tile<128, 8, SH128>(H1, p.n1W1t, acc);
    ln_gelu_to_lds<8>(acc, p.n1b1, p.n1g1, p.n1be1, H2, SH128); }
  __syncthreads();
  { v8f acc[16]; gemm_tile<128, 16, SH128>(H2, p.n1W2t, acc);
    const int hh = l >> 4, nl = l & 15;
#pragma unroll
    for (int r = 0; r < 8; ++r) {
      const int m2 = r + 8 * hh;
      float* dst = p.sums + (size_t)colArr[m2] * 256;
#pragma unroll
      for (int nt = 0; nt < 16; ++nt) {
        const int n = nt * 16 + nl;
        atomicAdd(dst + n, acc[nt][r] + p.n1b2[n]);
      }
    }
    if (l < 16) atomicAdd(p.cnt + colArr[l], 1.0f);
  }
}

// ---------------------------------------------------------------- node kernel

struct NodeParams {
  const int* c_ids; const float* concept;
  const float* sums; const float* cnt;
  const _Float16 *W0t, *W1t, *W2t;
  const float *b0, *g0, *be0, *b1, *g1, *be1, *b2;
  float* xOut;
};

__global__ __launch_bounds__(64) void node_kernel(NodeParams p) {
  __shared__ __align__(16) _Float16 sA [2][16 * SA384];
  __shared__ __align__(16) _Float16 sH1[2][16 * SH128];
  __shared__ __align__(16) _Float16 sH2[2][16 * SH128];

  const int w = threadIdx.x >> 5;
  const int l = threadIdx.x & 31;
  const int vBase = (blockIdx.x * 2 + w) * 16;
  _Float16* A  = sA[w];
  _Float16* H1 = sH1[w];
  _Float16* H2 = sH2[w];

  // ---- gather: A = [x | sums/max(cnt,1)] ; clamp OOB rows, guard stores ----
  {
    const int m = l >> 1, pp = l & 1;
    int vv = vBase + m; vv = (vv < NNODES) ? vv : (NNODES - 1);
    const float4* xr = (const float4*)(p.concept + (size_t)p.c_ids[vv] * CD);
    const float4* ag = (const float4*)(p.sums + (size_t)vv * 256);
    const float inv = 1.0f / fmaxf(p.cnt[vv], 1.0f);
#pragma unroll
    for (int i = 0; i < 16; ++i)
      store4h(A + m * SA384 + pp * 64 + i * 4, xr[pp * 16 + i]);
#pragma unroll
    for (int i = 0; i < 32; ++i) {
      float4 a4 = ag[pp * 32 + i];
      a4.x *= inv; a4.y *= inv; a4.z *= inv; a4.w *= inv;
      store4h(A + m * SA384 + 128 + pp * 128 + i * 4, a4);
    }
  }
  __syncthreads();

  { v8f acc[8]; gemm_tile<384, 8, SA384>(A, p.W0t, acc);
    ln_gelu_to_lds<8>(acc, p.b0, p.g0, p.be0, H1, SH128); }
  __syncthreads();
  { v8f acc[8]; gemm_tile<128, 8, SH128>(H1, p.W1t, acc);
    ln_gelu_to_lds<8>(acc, p.b1, p.g1, p.be1, H2, SH128); }
  __syncthreads();
  { v8f acc[8]; gemm_tile<128, 8, SH128>(H2, p.W2t, acc);
    const int hh = l >> 4, nl = l & 15;
#pragma unroll
    for (int nt = 0; nt < 8; ++nt) {
      const int n = nt * 16 + nl;
      const float bv = p.b2[n];
#pragma unroll
      for (int r = 0; r < 8; ++r) {
        const int vv = vBase + r + 8 * hh;
        if (vv < NNODES) p.xOut[(size_t)vv * CD + n] = acc[nt][r] + bv;
      }
    }
  }
}

// ---------------------------------------------------------------- launch

extern "C" void kernel_launch(void* const* d_in, const int* in_sizes, int n_in,
                              void* d_out, int out_size, void* d_ws, size_t ws_size,
                              hipStream_t stream) {
  auto F = [&](int i) { return (const float*)d_in[i]; };

  const int*   edge_index = (const int*)d_in[0];
  const int*   c_ids      = (const int*)d_in[1];
  const float* u          = F(2);
  const int*   rel_ids    = (const int*)d_in[5];
  const float* concept    = F(6);
  const float* rel_tab    = F(7);

  // MLP param pointers (10 arrays each). Handle both flattening conventions:
  // per-array entries (n_in == 38) or one concatenated buffer per tuple.
  const float *eP[10], *n1P[10], *n2P[10];
  if (n_in >= 38) {
    for (int i = 0; i < 10; ++i) { eP[i] = F(8 + i); n1P[i] = F(18 + i); n2P[i] = F(28 + i); }
  } else {
    const int szE[10]  = {384*128,128,128,128, 128*128,128,128,128, 128*128,128};
    const int szN1[10] = {256*128,128,128,128, 128*128,128,128,128, 128*256,256};
    const int szN2[10] = {384*128,128,128,128, 128*128,128,128,128, 128*128,128};
    const float* b = F(8);  for (int i = 0; i < 10; ++i) { eP[i]  = b; b += szE[i];  }
    b = F(9);               for (int i = 0; i < 10; ++i) { n1P[i] = b; b += szN1[i]; }
    b = F(10);              for (int i = 0; i < 10; ++i) { n2P[i] = b; b += szN2[i]; }
  }

  // Workspace layout
  char* ws = (char*)d_ws;
  float* sums = (float*)ws;                         // 25000*256 f32
  float* cnt  = (float*)(ws + (size_t)NNODES * 256 * 4);  // 25000 f32 (contiguous)
  _Float16* wt = (_Float16*)(ws + (size_t)NNODES * 256 * 4 + (size_t)NNODES * 4);
  _Float16* eW0t  = wt;                  // 128x384
  _Float16* eW1t  = eW0t  + 128 * 384;   // 128x128
  _Float16* eW2t  = eW1t  + 128 * 128;
  _Float16* n1W0t = eW2t  + 128 * 128;   // 128x256
  _Float16* n1W1t = n1W0t + 128 * 256;
  _Float16* n1W2t = n1W1t + 128 * 128;   // 256x128
  _Float16* n2W0t = n1W2t + 256 * 128;   // 128x384
  _Float16* n2W1t = n2W0t + 128 * 384;
  _Float16* n2W2t = n2W1t + 128 * 128;

  auto cvt = [&](const float* W, _Float16* Wt, int K, int N) {
    int tot = K * N;
    wt_convert<<<(tot + 255) / 256, 256, 0, stream>>>(W, Wt, K, N);
  };
  cvt(eP[0],  eW0t,  384, 128);
  cvt(eP[4],  eW1t,  128, 128);
  cvt(eP[8],  eW2t,  128, 128);
  cvt(n1P[0], n1W0t, 256, 128);
  cvt(n1P[4], n1W1t, 128, 128);
  cvt(n1P[8], n1W2t, 128, 256);
  cvt(n2P[0], n2W0t, 384, 128);
  cvt(n2P[4], n2W1t, 128, 128);
  cvt(n2P[8], n2W2t, 128, 128);

  // Zero accumulators each call (sums + cnt are contiguous).
  {
    int tot = NNODES * 256 + NNODES;
    zero_f32<<<(tot + 255) / 256, 256, 0, stream>>>(sums, tot);
  }

  float* outX  = (float*)d_out;                     // [25000,128]
  float* outEA = outX + (size_t)NNODES * CD;        // [400000,128]
  float* outU  = outEA + (size_t)NEDGES * CD;       // [16]

  EdgeParams ep;
  ep.edge_index = edge_index; ep.c_ids = c_ids; ep.rel_ids = rel_ids;
  ep.concept = concept; ep.rel = rel_tab;
  ep.eW0t = eW0t; ep.eW1t = eW1t; ep.eW2t = eW2t;
  ep.n1W0t = n1W0t; ep.n1W1t = n1W1t; ep.n1W2t = n1W2t;
  ep.eb0 = eP[1]; ep.eg0 = eP[2]; ep.ebe0 = eP[3];
  ep.eb1 = eP[5]; ep.eg1 = eP[6]; ep.ebe1 = eP[7]; ep.eb2 = eP[9];
  ep.n1b0 = n1P[1]; ep.n1g0 = n1P[2]; ep.n1be0 = n1P[3];
  ep.n1b1 = n1P[5]; ep.n1g1 = n1P[6]; ep.n1be1 = n1P[7]; ep.n1b2 = n1P[9];
  ep.eaOut = outEA; ep.sums = sums; ep.cnt = cnt;
  edge_kernel<<<(NEDGES / 16) / 2, 64, 0, stream>>>(ep);   // 12500 blocks

  NodeParams np;
  np.c_ids = c_ids; np.concept = concept; np.sums = sums; np.cnt = cnt;
  np.W0t = n2W0t; np.W1t = n2W1t; np.W2t = n2W2t;
  np.b0 = n2P[1]; np.g0 = n2P[2]; np.be0 = n2P[3];
  np.b1 = n2P[5]; np.g1 = n2P[6]; np.be1 = n2P[7]; np.b2 = n2P[9];
  np.xOut = outX;
  const int nodeTiles = (NNODES + 15) / 16;                // 1563
  node_kernel<<<(nodeTiles + 1) / 2, 64, 0, stream>>>(np); // 782 blocks

  copy_u<<<1, 32, 0, stream>>>(u, outU);
}
